// EquivariantConvLayer_65584150610131
// MI455X (gfx1250) — compile-verified, matrix operand order checked by reading, not run on hardware
//
#include <hip/hip_runtime.h>
#include <hip/hip_bf16.h>
#include <stdint.h>

// EquivariantConvLayer for MI455X (gfx1250, wave32, WMMA bf16)
//
// N=20000 nodes, E=640000 edges, D0=64 scalar ch, D1=32 vector ch, H=64.
// Heavy terms restructured as per-row-scaled WMMA accumulation:
//   msg0e = sum_p diag(s[:,p]) * (H @ w_ss[p])   -> 64 accum steps, K=64
//   msg1o = sum_d diag(v[:,d,x]) * (H @ w_vs[d]) -> 32 accum steps * 3 x
// Weights pre-swizzled into WMMA B-fragment-major bf16 and streamed to LDS
// with the Tensor Data Mover (tensor_load_to_lds / s_wait_tensorcnt).

#define D0C 64
#define D1C 32
#define HC  64

typedef __bf16        v16bf __attribute__((ext_vector_type(16)));
typedef float         v8f   __attribute__((ext_vector_type(8)));
typedef unsigned int  u32x4 __attribute__((ext_vector_type(4)));
typedef int           i32x8 __attribute__((ext_vector_type(8)));
typedef int           i32x4 __attribute__((ext_vector_type(4)));

union FragA {
  v16bf v;
  u32x4 q[2];
  unsigned u[8];
  __bf16 e[16];
};

__device__ __forceinline__ float silu_f(float x) { return x / (1.0f + __expf(-x)); }

__device__ __forceinline__ unsigned short f2bf(float x) {
  union { __bf16 h; unsigned short u; } c; c.h = (__bf16)x; return c.u;
}
__device__ __forceinline__ unsigned bf16dup(float x) {
  unsigned u = (unsigned)f2bf(x); return (u << 16) | u;
}

// Scale a bf16 16x32 A-fragment by a per-row scalar (rows are per-lane in the
// CDNA5 A layout, so a replicated scalar is correct) using v_pk_mul_bf16.
__device__ __forceinline__ v16bf scale_frag(v16bf h, unsigned sdup) {
  FragA a, r; a.v = h;
#pragma unroll
  for (int i = 0; i < 8; ++i)
    asm("v_pk_mul_bf16 %0, %1, %2" : "=v"(r.u[i]) : "v"(a.u[i]), "v"(sdup));
  return r.v;
}

__device__ __forceinline__ v8f wmma_bf16(v16bf a, v16bf b, v8f c) {
  return __builtin_amdgcn_wmma_f32_16x16x32_bf16(false, a, false, b,
                                                 (short)0, c, false, false);
}

// A-fragment K index for element i of lane (bf16 16x32 layout, ISA 7.12.2)
__device__ __forceinline__ int akidx(int lane, int i) {
  return ((lane >> 4) << 3) + ((i >> 3) << 4) + (i & 7);
}

// Stage nbytes (multiple of 16, tile fits 16-bit dim in 8B units) from global
// to LDS. Uses the Tensor Data Mover with the 6-arg builtin
// (g0, g1, g2, g3, g4-spare, cpol); cooperative copy fallback otherwise.
__device__ __forceinline__ void stage_lds(void* lds_dst, const void* gsrc,
                                          unsigned nbytes, int tid, int nthr) {
#if __has_builtin(__builtin_amdgcn_tensor_load_to_lds) && __has_builtin(__builtin_amdgcn_s_wait_tensorcnt)
  (void)nthr;
  if (tid < 32) {  // one wave issues the DMA (EXEC is ignored by TDM)
    unsigned long long ga = (unsigned long long)gsrc;
    unsigned lo = (unsigned)(unsigned long long)lds_dst;  // addr[31:0] == LDS offset
    unsigned n8 = nbytes >> 3;                            // elements of 8B
    u32x4 g0;
    g0[0] = 1u;                                  // count=1, user descriptor
    g0[1] = lo;                                  // lds_addr
    g0[2] = (unsigned)ga;                        // global_addr[31:0]
    g0[3] = (unsigned)(ga >> 32) | 0x80000000u;  // global_addr[56:32] | type=2
    i32x8 g1;
    g1[0] = (int)(3u << 16);                 // wg_mask=0, data_size=8B
    g1[1] = (int)((n8 & 0xFFFFu) << 16);     // tensor_dim0[15:0]
    g1[2] = (int)((n8 >> 16) | (1u << 16));  // tensor_dim0[31:16] | tensor_dim1=1
    g1[3] = (int)((n8 & 0xFFFFu) << 16);     // tile_dim0 = n8
    g1[4] = 0;                               // tile_dim1=0, tile_dim2=0
    g1[5] = (int)n8;                         // tensor_dim0_stride[31:0]
    g1[6] = 0;
    g1[7] = 0;
    i32x4 z4 = {0, 0, 0, 0};
    i32x8 z8 = {0, 0, 0, 0, 0, 0, 0, 0};
    __builtin_amdgcn_tensor_load_to_lds(g0, g1, z4, z4, z8, 0);
    __builtin_amdgcn_s_wait_tensorcnt((short)0);
  }
#else
  const u32x4* s = (const u32x4*)gsrc;
  u32x4* d = (u32x4*)lds_dst;
  unsigned n16 = nbytes >> 4;
  for (unsigned i = (unsigned)tid; i < n16; i += (unsigned)nthr) d[i] = s[i];
#endif
}

// Load the bf16 A-fragment of edge_scalar for one row (two 16B runs).
__device__ __forceinline__ v16bf load_hfrag(const unsigned short* h16,
                                            long erow, int ks, int lane) {
  int kbase = (lane >> 4) << 3;
  const u32x4* p0 = (const u32x4*)(h16 + erow * 64 + ks * 32 + kbase);
  const u32x4* p1 = (const u32x4*)(h16 + erow * 64 + ks * 32 + kbase + 16);
  FragA f; f.q[0] = p0[0]; f.q[1] = p1[0];
  return f.v;
}

// ---------------------------------------------------------------------------
// K0: zero aggregation buffers (aggr0 | aggr1 | counts are contiguous)
__global__ void k_zero(float* p, int n) {
  int i = blockIdx.x * 256 + threadIdx.x;
  if (i < n) p[i] = 0.0f;
}

// ---------------------------------------------------------------------------
// K1: convert + swizzle weights into WMMA B-fragment-major bf16 layout:
//   block(per 32x16 B tile) = [.. ][nt][ks], then [lane(32)][elem(16)]
__global__ void k_prep(const float* W1, const float* W2, const float* W3,
                       const float* w_ss, const float* w_vs,
                       unsigned short* wmlp, unsigned short* wss,
                       unsigned short* wvs) {
  const int NM = 3 * 4 * 2 * 32 * 16;    // 12288  (W1,W2,W3)
  const int NSS = 64 * 4 * 2 * 32 * 16;  // 262144 (w_ss)
  const int NVS = 32 * 2 * 2 * 32 * 16;  // 65536  (w_vs)
  int t = blockIdx.x * 256 + threadIdx.x;
  if (t < NM) {
    int lyr = t / 4096, r = t % 4096;
    int blk = r >> 9, lane = (r >> 4) & 31, i = r & 15;
    int nt = blk >> 1, ks = blk & 1;
    int k = ks * 32 + akidx(lane, i);
    int n = nt * 16 + (lane & 15);
    const float* W = (lyr == 0) ? W1 : (lyr == 1 ? W2 : W3);
    wmlp[t] = f2bf(W[k * 64 + n]);
  } else if (t < NM + NSS) {
    int t2 = t - NM;
    int p = t2 >> 12, r = t2 & 4095;
    int blk = r >> 9, lane = (r >> 4) & 31, i = r & 15;
    int nt = blk >> 1, ks = blk & 1;
    int k = ks * 32 + akidx(lane, i);
    int n = nt * 16 + (lane & 15);
    wss[t2] = f2bf(w_ss[(p * 64 + k) * 64 + n]);
  } else if (t < NM + NSS + NVS) {
    int t3 = t - NM - NSS;
    int d = t3 >> 11, r = t3 & 2047;
    int blk = r >> 9, lane = (r >> 4) & 31, i = r & 15;
    int nt = blk >> 1, ks = blk & 1;
    int k = ks * 32 + akidx(lane, i);
    int n = nt * 16 + (lane & 15);
    wvs[t3] = f2bf(w_vs[(d * 64 + k) * 32 + n]);
  }
}

// ---------------------------------------------------------------------------
// K2: edge MLP (3x 64x64 + silu) via WMMA. 4 waves/WG, 16 edges/wave.
// Writes edge_scalar as bf16 row-major and accumulates degree counts.
__global__ void __launch_bounds__(128)
k_edge_mlp(const float* ea, const float* b1, const float* b2, const float* b3,
           const int* eidx, const unsigned short* wmlp_g,
           unsigned short* h16, float* counts, int E) {
  __shared__ alignas(16) unsigned char wmlp[24576];  // 3 layers x 8KB frags
  __shared__ float act[4][16][68];
  int tid = threadIdx.x, w = tid >> 5, lane = tid & 31;
  int row = lane & 15, half = lane >> 4;
  long e0 = (long)blockIdx.x * 64 + w * 16;

  stage_lds(wmlp, wmlp_g, 24576, tid, 128);

  const float* ap = ea + (e0 + row) * 64 + half * 32;
#pragma unroll
  for (int j = 0; j < 8; ++j) {
    float4 v = ((const float4*)ap)[j];
    act[w][row][half * 32 + j * 4 + 0] = v.x;
    act[w][row][half * 32 + j * 4 + 1] = v.y;
    act[w][row][half * 32 + j * 4 + 2] = v.z;
    act[w][row][half * 32 + j * 4 + 3] = v.w;
  }
  if (lane < 16) atomicAdd(&counts[eidx[E + e0 + lane]], 1.0f);
  __syncthreads();

#pragma unroll
  for (int lyr = 0; lyr < 3; ++lyr) {
    const float* bias = (lyr == 0) ? b1 : (lyr == 1 ? b2 : b3);
    v16bf af[2];
#pragma unroll
    for (int ks = 0; ks < 2; ++ks) {
      FragA u;
#pragma unroll
      for (int i = 0; i < 16; ++i)
        u.e[i] = (__bf16)act[w][row][ks * 32 + akidx(lane, i)];
      af[ks] = u.v;
    }
    float res[4][8];
#pragma unroll
    for (int nt = 0; nt < 4; ++nt) {
      v8f c = {0, 0, 0, 0, 0, 0, 0, 0};
#pragma unroll
      for (int ks = 0; ks < 2; ++ks) {
        FragA b;
        const u32x4* bp =
            (const u32x4*)(wmlp + ((lyr * 4 + nt) * 2 + ks) * 1024 + lane * 32);
        b.q[0] = bp[0]; b.q[1] = bp[1];
        c = wmma_bf16(af[ks], b.v, c);
      }
      float bv = bias[nt * 16 + (lane & 15)];
#pragma unroll
      for (int j = 0; j < 8; ++j) {
        float x = c[j] + bv;
        res[nt][j] = (lyr < 2) ? silu_f(x) : x;
      }
    }
#pragma unroll
    for (int nt = 0; nt < 4; ++nt)
#pragma unroll
      for (int j = 0; j < 8; ++j) {
        int m = (lane >> 4) * 8 + j;
        int n = nt * 16 + (lane & 15);
        if (lyr < 2) act[w][m][n] = res[nt][j];
        else h16[(e0 + m) * 64 + n] = f2bf(res[nt][j]);
      }
  }
}

// ---------------------------------------------------------------------------
// K3: msg0e. 4 waves/WG (wave = one 16-col N tile), 64 edges/WG.
// acc(e,o) += sum_p s(e,p) * (h(e,:) @ w_ss[p])(o)  +  dot_vv @ w_vv (VALU)
__global__ void __launch_bounds__(128)
k_msg0(const float* feat0e, const float* feat1o, const float* pos,
       const int* eidx, const unsigned short* h16, const unsigned short* wss_g,
       const float* w_vv, float* aggr0, int E) {
  __shared__ alignas(16) unsigned char wbuf[65536];  // 8 p-slices of w_ss frags
  __shared__ float s_tile[64][68];
  __shared__ float dd[64][33];
  __shared__ int srcb[64], dstb[64];
  int tid = threadIdx.x, wv = tid >> 5, lane = tid & 31;
  int nt = wv;
  long e0 = (long)blockIdx.x * 64;

  if (tid < 64) {
    srcb[tid] = eidx[e0 + tid];
    dstb[tid] = eidx[E + e0 + tid];
  }
  __syncthreads();

  for (int idx = tid; idx < 64 * 64; idx += 128) {
    int e = idx >> 6, c = idx & 63;
    s_tile[e][c] = feat0e[(long)srcb[e] * 64 + c];
  }
  {
    int e = tid >> 1, dh = (tid & 1) * 16;
    int si = srcb[e], di = dstb[e];
    float rx = pos[di * 3 + 0] - pos[si * 3 + 0];
    float ry = pos[di * 3 + 1] - pos[si * 3 + 1];
    float rz = pos[di * 3 + 2] - pos[si * 3 + 2];
    const float* vp = feat1o + (long)si * 96;
#pragma unroll
    for (int d = 0; d < 16; ++d) {
      int dg = dh + d;
      dd[e][dg] = vp[dg * 3 + 0] * rx + vp[dg * 3 + 1] * ry + vp[dg * 3 + 2] * rz;
    }
  }

  v16bf hf[4][2];
#pragma unroll
  for (int mt = 0; mt < 4; ++mt)
#pragma unroll
    for (int ks = 0; ks < 2; ++ks)
      hf[mt][ks] = load_hfrag(h16, e0 + mt * 16 + (lane & 15), ks, lane);

  v8f acc[4];
#pragma unroll
  for (int mt = 0; mt < 4; ++mt) acc[mt] = (v8f){0, 0, 0, 0, 0, 0, 0, 0};

  for (int pc = 0; pc < 8; ++pc) {
    __syncthreads();
    stage_lds(wbuf, wss_g + (size_t)pc * 8 * 4096, 65536, tid, 128);
    __syncthreads();
#pragma unroll
    for (int pl = 0; pl < 8; ++pl) {
      FragA b0, b1;
      const u32x4* bp0 =
          (const u32x4*)(wbuf + ((pl * 4 + nt) * 2 + 0) * 1024 + lane * 32);
      const u32x4* bp1 =
          (const u32x4*)(wbuf + ((pl * 4 + nt) * 2 + 1) * 1024 + lane * 32);
      b0.q[0] = bp0[0]; b0.q[1] = bp0[1];
      b1.q[0] = bp1[0]; b1.q[1] = bp1[1];
      int p = pc * 8 + pl;
#pragma unroll
      for (int mt = 0; mt < 4; ++mt) {
        unsigned sdup = bf16dup(s_tile[mt * 16 + (lane & 15)][p]);
        acc[mt] = wmma_bf16(scale_frag(hf[mt][0], sdup), b0.v, acc[mt]);
        acc[mt] = wmma_bf16(scale_frag(hf[mt][1], sdup), b1.v, acc[mt]);
      }
    }
  }

  int n = nt * 16 + (lane & 15);
  float wvv[32];
#pragma unroll
  for (int d = 0; d < 32; ++d) wvv[d] = w_vv[d * 64 + n];
#pragma unroll
  for (int mt = 0; mt < 4; ++mt)
#pragma unroll
    for (int j = 0; j < 8; ++j) {
      int el = mt * 16 + (lane >> 4) * 8 + j;
      float val = acc[mt][j];
#pragma unroll
      for (int d = 0; d < 32; ++d) val += dd[el][d] * wvv[d];
      atomicAdd(&aggr0[(long)dstb[el] * 64 + n], val);
    }
}

// ---------------------------------------------------------------------------
// K4: msg1o. 6 waves/WG: wave -> (x = w/2, ntile = w&1), 64 edges/WG.
// acc(e,c,x) += sum_d v(e,d,x) * (h(e,:) @ w_vs[d])(c)
//            +  sv(e,c)*rel(e,x)  +  cross(e,:,x) @ w_cross (VALU)
__global__ void __launch_bounds__(192)
k_msg1(const float* feat0e, const float* feat1o, const float* pos,
       const int* eidx, const unsigned short* h16, const unsigned short* wvs_g,
       const float* w_sv, const float* w_cross, float* aggr1, int E) {
  __shared__ alignas(16) unsigned char wbuf[65536];  // 16 d-slices of w_vs
  __shared__ float v_tile[64][3][33];
  __shared__ float cr[64][3][33];
  __shared__ float sv[64][33];
  __shared__ float rel[64][4];
  __shared__ int srcb[64], dstb[64];
  int tid = threadIdx.x, wv = tid >> 5, lane = tid & 31;
  int x = wv >> 1, nt = wv & 1;
  long e0 = (long)blockIdx.x * 64;

  if (tid < 64) {
    int si = eidx[e0 + tid], di = eidx[E + e0 + tid];
    srcb[tid] = si; dstb[tid] = di;
    rel[tid][0] = pos[di * 3 + 0] - pos[si * 3 + 0];
    rel[tid][1] = pos[di * 3 + 1] - pos[si * 3 + 1];
    rel[tid][2] = pos[di * 3 + 2] - pos[si * 3 + 2];
  }
  __syncthreads();

  for (int idx = tid; idx < 64 * 96; idx += 192) {
    int e = idx / 96, r = idx % 96;
    v_tile[e][r % 3][r / 3] = feat1o[(long)srcb[e] * 96 + r];
  }
  for (int idx = tid; idx < 2048; idx += 192) {
    int e = idx >> 5, c = idx & 31;
    const float* sp = feat0e + (long)srcb[e] * 64;
    float a = 0.f;
#pragma unroll 8
    for (int p = 0; p < 64; ++p) a += sp[p] * w_sv[p * 32 + c];
    sv[e][c] = a;
  }
  for (int idx = tid; idx < 2048; idx += 192) {
    int e = idx >> 5, d = idx & 31;
    const float* vp = feat1o + (long)srcb[e] * 96 + d * 3;
    float vx = vp[0], vy = vp[1], vz = vp[2];
    float rx = rel[e][0], ry = rel[e][1], rz = rel[e][2];
    cr[e][0][d] = vy * rz - vz * ry;
    cr[e][1][d] = vz * rx - vx * rz;
    cr[e][2][d] = vx * ry - vy * rx;
  }

  v16bf hf[4][2];
#pragma unroll
  for (int mt = 0; mt < 4; ++mt)
#pragma unroll
    for (int ks = 0; ks < 2; ++ks)
      hf[mt][ks] = load_hfrag(h16, e0 + mt * 16 + (lane & 15), ks, lane);

  v8f acc[4];
#pragma unroll
  for (int mt = 0; mt < 4; ++mt) acc[mt] = (v8f){0, 0, 0, 0, 0, 0, 0, 0};

  for (int dc = 0; dc < 2; ++dc) {
    __syncthreads();
    stage_lds(wbuf, wvs_g + (size_t)dc * 16 * 2048, 65536, tid, 192);
    __syncthreads();
#pragma unroll
    for (int dl = 0; dl < 16; ++dl) {
      FragA b0, b1;
      const u32x4* bp0 =
          (const u32x4*)(wbuf + ((dl * 2 + nt) * 2 + 0) * 1024 + lane * 32);
      const u32x4* bp1 =
          (const u32x4*)(wbuf + ((dl * 2 + nt) * 2 + 1) * 1024 + lane * 32);
      b0.q[0] = bp0[0]; b0.q[1] = bp0[1];
      b1.q[0] = bp1[0]; b1.q[1] = bp1[1];
      int dg = dc * 16 + dl;
#pragma unroll
      for (int mt = 0; mt < 4; ++mt) {
        unsigned sdup = bf16dup(v_tile[mt * 16 + (lane & 15)][x][dg]);
        acc[mt] = wmma_bf16(scale_frag(hf[mt][0], sdup), b0.v, acc[mt]);
        acc[mt] = wmma_bf16(scale_frag(hf[mt][1], sdup), b1.v, acc[mt]);
      }
    }
  }

  int n = nt * 16 + (lane & 15);
  float wcr[32];
#pragma unroll
  for (int d = 0; d < 32; ++d) wcr[d] = w_cross[d * 32 + n];
#pragma unroll
  for (int mt = 0; mt < 4; ++mt)
#pragma unroll
    for (int j = 0; j < 8; ++j) {
      int el = mt * 16 + (lane >> 4) * 8 + j;
      float val = acc[mt][j] + sv[el][n] * rel[el][x];
#pragma unroll
      for (int d = 0; d < 32; ++d) val += cr[el][x][d] * wcr[d];
      atomicAdd(&aggr1[((long)dstb[el] * 32 + n) * 3 + x], val);
    }
}

// ---------------------------------------------------------------------------
// K5: per-node finalize: mean, silu, gates, residual, concat output [N,160]
__global__ void k_final(const float* aggr0, const float* aggr1,
                        const float* counts, const float* feat0e,
                        const float* feat1o, const float* Wg, const float* bg,
                        float* out, int N) {
  int n = blockIdx.x * 256 + threadIdx.x;
  if (n >= N) return;
  float inv = 1.0f / fmaxf(counts[n], 1.0f);
  float a0[64];
#pragma unroll 8
  for (int o = 0; o < 64; ++o) a0[o] = aggr0[(long)n * 64 + o] * inv;
#pragma unroll 8
  for (int o = 0; o < 64; ++o)
    out[(long)n * 160 + o] = silu_f(a0[o]) + feat0e[(long)n * 64 + o];
  for (int c = 0; c < 32; ++c) {
    float g = bg[c];
#pragma unroll 8
    for (int o = 0; o < 64; ++o) g += a0[o] * Wg[o * 32 + c];
    g = silu_f(g);
#pragma unroll
    for (int xx = 0; xx < 3; ++xx) {
      long i1 = ((long)n * 32 + c) * 3 + xx;
      out[(long)n * 160 + 64 + c * 3 + xx] = aggr1[i1] * inv * g + feat1o[i1];
    }
  }
}

// ---------------------------------------------------------------------------
extern "C" void kernel_launch(void* const* d_in, const int* in_sizes, int n_in,
                              void* d_out, int out_size, void* d_ws,
                              size_t ws_size, hipStream_t stream) {
  const float* feat0e  = (const float*)d_in[0];
  const float* feat1o  = (const float*)d_in[1];
  const float* ea      = (const float*)d_in[2];
  const float* pos     = (const float*)d_in[3];
  const int*   eidx    = (const int*)d_in[4];
  const float* W1      = (const float*)d_in[5];
  const float* b1      = (const float*)d_in[6];
  const float* W2      = (const float*)d_in[7];
  const float* b2      = (const float*)d_in[8];
  const float* W3      = (const float*)d_in[9];
  const float* b3      = (const float*)d_in[10];
  const float* w_ss    = (const float*)d_in[11];
  const float* w_vv    = (const float*)d_in[12];
  const float* w_sv    = (const float*)d_in[13];
  const float* w_vs    = (const float*)d_in[14];
  const float* w_cross = (const float*)d_in[15];
  const float* Wg      = (const float*)d_in[16];
  const float* bg      = (const float*)d_in[17];
  (void)n_in; (void)out_size; (void)ws_size;

  int N = in_sizes[0] / D0C;        // 20000
  int E = in_sizes[2] / HC;         // 640000

  char* ws = (char*)d_ws;
  size_t off = 0;
  unsigned short* h16 = (unsigned short*)(ws + off);  off += (size_t)E * 64 * 2;
  float* aggr0  = (float*)(ws + off);                 off += (size_t)N * 64 * 4;
  float* aggr1  = (float*)(ws + off);                 off += (size_t)N * 96 * 4;
  float* counts = (float*)(ws + off);                 off += (size_t)N * 4;
  off = (off + 255) & ~(size_t)255;
  unsigned short* wmlp = (unsigned short*)(ws + off); off += 12288 * 2;
  off = (off + 255) & ~(size_t)255;
  unsigned short* wss  = (unsigned short*)(ws + off); off += 262144 * 2;
  off = (off + 255) & ~(size_t)255;
  unsigned short* wvs  = (unsigned short*)(ws + off); off += 65536 * 2;

  int nz = N * (64 + 96 + 1);  // aggr0 | aggr1 | counts are contiguous
  hipLaunchKernelGGL(k_zero, dim3((nz + 255) / 256), dim3(256), 0, stream,
                     aggr0, nz);

  int nswz = 12288 + 262144 + 65536;
  hipLaunchKernelGGL(k_prep, dim3((nswz + 255) / 256), dim3(256), 0, stream,
                     W1, W2, W3, w_ss, w_vs, wmlp, wss, wvs);

  hipLaunchKernelGGL(k_edge_mlp, dim3(E / 64), dim3(128), 0, stream,
                     ea, b1, b2, b3, eidx, wmlp, h16, counts, E);

  hipLaunchKernelGGL(k_msg0, dim3(E / 64), dim3(128), 0, stream,
                     feat0e, feat1o, pos, eidx, h16, wss, w_vv, aggr0, E);

  hipLaunchKernelGGL(k_msg1, dim3(E / 64), dim3(192), 0, stream,
                     feat0e, feat1o, pos, eidx, h16, wvs, w_sv, w_cross, aggr1,
                     E);

  hipLaunchKernelGGL(k_final, dim3((N + 255) / 256), dim3(256), 0, stream,
                     aggr0, aggr1, counts, feat0e, feat1o, Wg, bg,
                     (float*)d_out, N);
}
